// ValNetDense_90512140796482
// MI455X (gfx1250) — matrix operand order repaired and checked by reading, not compiled, not run on hardware
//
#include <hip/hip_runtime.h>
#include <hip/hip_bf16.h>

typedef _Float16 v16h __attribute__((ext_vector_type(16)));
typedef _Float16 v8h  __attribute__((ext_vector_type(8)));
typedef float    v8f  __attribute__((ext_vector_type(8)));
typedef float    v4f  __attribute__((ext_vector_type(4)));

#define TB 256
#define BMT 128
#define BNT 128
#define BKT 32
#define SAS 40   // LDS row stride in halves (conflict-reducing pad)

// ------------------------------------------------------------------
// WMMA GEMM: C[b] = act(A[b] * B[b] + bias), optional per-row mask.
// A: f32 or f16, optionally transposed (element (m,k) at A[k*lda+m]).
// B: f16 row-major [K,N], C: f16 row-major.
// Pipelined: next K-tile is fetched into registers during compute.
// ------------------------------------------------------------------
template<bool AF32, bool TRA, bool RELU, bool MASK>
__global__ __launch_bounds__(TB)
void gemm_wmma(const void* __restrict__ Ap, const _Float16* __restrict__ Bp,
               const _Float16* __restrict__ biasp, const int* __restrict__ maskp,
               _Float16* __restrict__ Cp,
               int M, int N, int K,
               int lda, int ldb, int ldc,
               long strideA, long strideB, long strideC, long strideMask)
{
  __shared__ alignas(16) _Float16 As[BMT * SAS];
  __shared__ alignas(16) _Float16 Bs[BNT * SAS];

  const int b = blockIdx.z;
  const float*    Af = (const float*)Ap + (AF32 ? strideA * b : 0);
  const _Float16* Ah = (const _Float16*)Ap + (AF32 ? 0 : strideA * b);
  const _Float16* Bb = Bp + strideB * b;
  _Float16* Cb = Cp + strideC * b;
  const int* mb = MASK ? (maskp + strideMask * b) : nullptr;

  const int tile_m = blockIdx.y * BMT;
  const int tile_n = blockIdx.x * BNT;
  const int tid  = threadIdx.x;
  const int lane = tid & 31;
  const int wid  = tid >> 5;
  const int wm = (wid >> 2) * 64;   // 0 / 64
  const int wn = (wid & 3) * 32;    // 0..96
  const int l15 = lane & 15;
  const int lhi = lane >> 4;        // 0 / 1

  // staging geometry (16 elements / thread / tile)
  const int arow = TRA ? (tid >> 3) : (tid >> 1);              // TRA: k ; else m-row
  const int acol = TRA ? ((tid & 7) * 16) : ((tid & 1) * 16);  // TRA: m0 ; else k-off
  const int bk   = tid >> 3;
  const int bn0  = (tid & 7) * 16;
  const bool aRowsFull = (tile_m + BMT) <= M;
  const bool bColsFull = (tile_n + BNT) <= N;

  auto loadA = [&](int k0, v8h r[2]) {
    const bool kFull = (k0 + BKT) <= K;
    if (!TRA) {
      const long gm   = (long)tile_m + arow;
      const long base = gm * (long)lda + k0 + acol;
      if (aRowsFull && kFull) {
        if (AF32) {
          const v4f* s = (const v4f*)(Af + base);
          const v4f f0 = s[0], f1 = s[1], f2 = s[2], f3 = s[3];
          #pragma unroll
          for (int j = 0; j < 4; ++j) {
            r[0][j] = (_Float16)f0[j]; r[0][4 + j] = (_Float16)f1[j];
            r[1][j] = (_Float16)f2[j]; r[1][4 + j] = (_Float16)f3[j];
          }
        } else {
          const v8h* s = (const v8h*)(Ah + base);
          r[0] = s[0]; r[1] = s[1];
        }
      } else {
        for (int j = 0; j < 16; ++j) {
          _Float16 v = (_Float16)0.f;
          if (gm < M && (k0 + acol + j) < K)
            v = AF32 ? (_Float16)Af[base + j] : Ah[base + j];
          r[j >> 3][j & 7] = v;
        }
      }
    } else {
      const long base = (long)(k0 + arow) * lda + tile_m + acol;
      if (aRowsFull && kFull) {
        if (AF32) {
          const v4f* s = (const v4f*)(Af + base);
          const v4f f0 = s[0], f1 = s[1], f2 = s[2], f3 = s[3];
          #pragma unroll
          for (int j = 0; j < 4; ++j) {
            r[0][j] = (_Float16)f0[j]; r[0][4 + j] = (_Float16)f1[j];
            r[1][j] = (_Float16)f2[j]; r[1][4 + j] = (_Float16)f3[j];
          }
        } else {
          const v8h* s = (const v8h*)(Ah + base);
          r[0] = s[0]; r[1] = s[1];
        }
      } else {
        for (int j = 0; j < 16; ++j) {
          _Float16 v = (_Float16)0.f;
          if ((k0 + arow) < K && (tile_m + acol + j) < M)
            v = AF32 ? (_Float16)Af[base + j] : Ah[base + j];
          r[j >> 3][j & 7] = v;
        }
      }
    }
  };

  auto loadB = [&](int k0, v8h r[2]) {
    const bool kFull = (k0 + BKT) <= K;
    const long base = (long)(k0 + bk) * ldb + tile_n + bn0;
    if (bColsFull && kFull) {
      const v8h* s = (const v8h*)(Bb + base);
      r[0] = s[0]; r[1] = s[1];
    } else {
      for (int j = 0; j < 16; ++j) {
        _Float16 v = (_Float16)0.f;
        if ((k0 + bk) < K && (tile_n + bn0 + j) < N)
          v = Bb[base + j];
        r[j >> 3][j & 7] = v;
      }
    }
  };

  auto storeA = [&](const v8h r[2]) {
    if (!TRA) {
      v8h* d = (v8h*)&As[arow * SAS + acol];
      d[0] = r[0]; d[1] = r[1];
    } else {
      #pragma unroll
      for (int j = 0; j < 16; ++j)
        As[(acol + j) * SAS + arow] = r[j >> 3][j & 7];
    }
  };

  auto storeB = [&](const v8h r[2]) {
    #pragma unroll
    for (int j = 0; j < 16; ++j)
      Bs[(bn0 + j) * SAS + bk] = r[j >> 3][j & 7];
  };

  v8f acc[4][2] = {};
  v8h ra[2], rb[2], na[2], nb[2];
  loadA(0, ra);
  loadB(0, rb);

  for (int k0 = 0; k0 < K; k0 += BKT) {
    storeA(ra);
    storeB(rb);
    __syncthreads();

    const int kn = k0 + BKT;
    if (kn < K) { loadA(kn, na); loadB(kn, nb); }

    #pragma unroll
    for (int i = 0; i < 4; ++i) {
      // A frag: lane<16 holds K {0..7,16..23}; lane>=16 holds K {8..15,24..31}
      const _Float16* pa = &As[(wm + i * 16 + l15) * SAS + lhi * 8];
      const v8h alo = *(const v8h*)pa;
      const v8h ahi = *(const v8h*)(pa + 16);
      v16h a;
      #pragma unroll
      for (int j = 0; j < 8; ++j) { a[j] = alo[j]; a[8 + j] = ahi[j]; }
      #pragma unroll
      for (int jn = 0; jn < 2; ++jn) {
        // B frag: lane<16 holds K {0..7,16..23}; lane>=16 holds K {8..15,24..31}, N = lane&15
        const _Float16* pb = &Bs[(wn + jn * 16 + l15) * SAS + lhi * 8];
        const v8h blo = *(const v8h*)pb;
        const v8h bhi = *(const v8h*)(pb + 16);
        v16h bm;
        #pragma unroll
        for (int j = 0; j < 8; ++j) { bm[j] = blo[j]; bm[8 + j] = bhi[j]; }
        acc[i][jn] = __builtin_amdgcn_wmma_f32_16x16x32_f16(
            false, a, false, bm, (short)0, acc[i][jn], false, false);
      }
    }
    __syncthreads();
    ra[0] = na[0]; ra[1] = na[1];
    rb[0] = nb[0]; rb[1] = nb[1];
  }

  // ---- epilogue: bias + relu + mask, f32 -> f16 store ----
  #pragma unroll
  for (int i = 0; i < 4; ++i) {
    const int mbase = tile_m + wm + i * 16 + lhi * 8;
    #pragma unroll
    for (int jn = 0; jn < 2; ++jn) {
      const int nn = tile_n + wn + jn * 16 + l15;
      if (nn >= N) continue;
      const float bv = biasp ? (float)biasp[nn] : 0.f;
      #pragma unroll
      for (int j = 0; j < 8; ++j) {
        const int mm = mbase + j;
        if (mm < M) {
          float v = acc[i][jn][j] + bv;
          if (RELU) v = v > 0.f ? v : 0.f;
          if (MASK) v = mb[mm] ? v : 0.f;
          Cb[(long)mm * ldc + nn] = (_Float16)v;
        }
      }
    }
  }
}

// ---------------- small helper kernels ----------------
__global__ void cvt_f32_f16(const float* __restrict__ in, _Float16* __restrict__ out, int n) {
  int i = blockIdx.x * blockDim.x + threadIdx.x;
  if (i < n) out[i] = (_Float16)in[i];
}

__global__ void encode_nodes(const int* __restrict__ Va, _Float16* __restrict__ valv,
                             _Float16* __restrict__ Xv, int NVc, int ldv, int ldxv) {
  int idx = blockIdx.x * blockDim.x + threadIdx.x; // b*NV + v
  if (idx >= 32 * NVc) return;
  int b = idx / NVc, v = idx - b * NVc;
  const int* base = Va + (long)b * 5 * NVc;
  int t0 = base[0 * NVc + v] + 1;
  int t1 = base[1 * NVc + v] + 1;
  float x = (float)base[2 * NVc + v] / 32.0f - 1.0f;
  float y = (float)base[3 * NVc + v] / 32.0f - 1.0f;
  int t4 = base[4 * NVc + v];
  _Float16 f[24];
  #pragma unroll
  for (int j = 0; j < 8; ++j) f[j]     = (_Float16)(j == t0 ? 1.f : 0.f);
  #pragma unroll
  for (int j = 0; j < 8; ++j) f[8 + j] = (_Float16)(j == t1 ? 1.f : 0.f);
  f[16] = (_Float16)x; f[17] = (_Float16)y;
  #pragma unroll
  for (int j = 0; j < 6; ++j) f[18 + j] = (_Float16)(j == t4 ? 1.f : 0.f);
  #pragma unroll
  for (int j = 0; j < 24; ++j) {
    valv[(long)idx * ldv + j] = f[j];
    Xv[(long)idx * ldxv + j]  = f[j];
  }
}

__global__ void encode_edges(const int* __restrict__ Ea, _Float16* __restrict__ Xe,
                             int NEc, int ldx) {
  int idx = blockIdx.x * blockDim.x + threadIdx.x;
  if (idx >= 32 * NEc) return;
  int t = Ea[idx];
  #pragma unroll
  for (int j = 0; j < 6; ++j) Xe[(long)idx * ldx + j] = (_Float16)(j == t ? 1.f : 0.f);
}

// vectorized column-block copy: w, ofs, lsrc, ldst all multiples of 8
__global__ void copy_cols8(const _Float16* __restrict__ src, _Float16* __restrict__ dst,
                           long rows, int w, int lsrc, int ldst, int ofs) {
  long i = blockIdx.x * (long)blockDim.x + threadIdx.x;
  int wc = w >> 3;
  if (i >= rows * (long)wc) return;
  long r = i / wc; int c = (int)(i - r * wc) << 3;
  *(v8h*)(dst + r * (long)ldst + ofs + c) = *(const v8h*)(src + r * (long)lsrc + c);
}

__global__ void bcast_u8(const _Float16* __restrict__ u, _Float16* __restrict__ dst,
                         int rowsPerB, int ldd, int ofs) {
  long i = blockIdx.x * (long)blockDim.x + threadIdx.x;
  long total = (long)32 * rowsPerB * 16;   // 128/8 chunks
  if (i >= total) return;
  int c = (int)(i & 15) << 3;
  long r = i >> 4;
  int b = (int)(r / rowsPerB);
  *(v8h*)(dst + r * (long)ldd + ofs + c) = *(const v8h*)(u + b * 128 + c);
}

__global__ void colsum(const _Float16* __restrict__ src, _Float16* __restrict__ dst,
                       int R, int C, int ldd, int ofs) {
  int b = blockIdx.x;
  int c = threadIdx.x;
  if (c >= C) return;
  float s = 0.f;
  const _Float16* p = src + (long)b * R * C + c;
  for (int r = 0; r < R; ++r) s += (float)p[(long)r * C];
  dst[(long)b * ldd + ofs + c] = (_Float16)s;
}

__global__ void final_out(const _Float16* __restrict__ u, const float* __restrict__ Wout,
                          const float* __restrict__ bout, float* __restrict__ out) {
  __shared__ float red[128];
  int b = blockIdx.x, c = threadIdx.x;
  red[c] = (float)u[b * 128 + c] * Wout[c];
  __syncthreads();
  for (int s = 64; s > 0; s >>= 1) {
    if (c < s) red[c] += red[c + s];
    __syncthreads();
  }
  if (c == 0) out[b] = red[0] + bout[0];
}

// ---------------- host orchestration ----------------
extern "C" void kernel_launch(void* const* d_in, const int* in_sizes, int n_in,
                              void* d_out, int out_size, void* d_ws, size_t ws_size,
                              hipStream_t stream) {
  (void)in_sizes; (void)out_size; (void)ws_size;
  if (n_in < 128) return;
  const int B = 32, NV = 512, NE = 2048;

  const int*   Va     = (const int*)d_in[0];
  const int*   Ea     = (const int*)d_in[1];
  const float* Es     = (const float*)d_in[2];
  const float* Er     = (const float*)d_in[3];
  const int*   mask_v = (const int*)d_in[4];
  const int*   mask_e = (const int*)d_in[5];
  const float* Wout   = (const float*)d_in[126];
  const float* bout   = (const float*)d_in[127];
  float* out = (float*)d_out;

  // MLP layer dims per block / mlp(0=fe,1=fv,2=fu)
  int dims[4][3][6];
  for (int blk = 0; blk < 4; ++blk) {
    int iu = blk ? 128 : 0, iv = blk ? 128 : 24, ie = blk ? 128 : 6;
    int fins[3] = { ie + 2 * iv + iu, iv + 128 + iu, iu + 256 };
    for (int m = 0; m < 3; ++m) {
      dims[blk][m][0] = fins[m];
      for (int l = 1; l <= 4; ++l) dims[blk][m][l] = 256;
      dims[blk][m][5] = 128;
    }
  }
  // f16 weight/bias offsets in workspace
  size_t woff[4][3][5], boff[4][3][5], wtot = 0;
  for (int blk = 0; blk < 4; ++blk)
    for (int m = 0; m < 3; ++m)
      for (int l = 0; l < 5; ++l) {
        woff[blk][m][l] = wtot; wtot += (size_t)dims[blk][m][l] * dims[blk][m][l + 1];
        boff[blk][m][l] = wtot; wtot += (size_t)dims[blk][m][l + 1];
      }

  // workspace carve (element offsets in f16 units, 256B aligned)
  size_t o = 0;
  auto take = [&](size_t elems) { size_t r = o; o += (elems + 127) & ~(size_t)127; return r; };
  _Float16* ws16 = (_Float16*)d_ws;
  _Float16* Wb    = ws16 + take(wtot);
  _Float16* pValv = ws16 + take((size_t)B * NV * 128);
  _Float16* pVale = ws16 + take((size_t)B * NE * 128);
  _Float16* pXe   = ws16 + take((size_t)B * NE * 512);   // also fe ping-pong buffer (alias)
  _Float16* pHe0  = ws16 + take((size_t)B * NE * 256);
  _Float16* pXv   = ws16 + take((size_t)B * NV * 384);   // also fv ping-pong buffer (alias)
  _Float16* pHv0  = ws16 + take((size_t)B * NV * 256);
  _Float16* pXu   = ws16 + take((size_t)32 * 384);
  _Float16* pHu0  = ws16 + take((size_t)32 * 256);
  _Float16* pHu1  = ws16 + take((size_t)32 * 256);
  _Float16* pU    = ws16 + take((size_t)32 * 128);

  // convert all MLP weights/biases to f16
  {
    int ii = 6;
    for (int blk = 0; blk < 4; ++blk)
      for (int m = 0; m < 3; ++m)
        for (int l = 0; l < 5; ++l) {
          const float* Wp = (const float*)d_in[ii++];
          const float* bp = (const float*)d_in[ii++];
          int nw = dims[blk][m][l] * dims[blk][m][l + 1];
          int nb = dims[blk][m][l + 1];
          cvt_f32_f16<<<(nw + 255) / 256, 256, 0, stream>>>(Wp, Wb + woff[blk][m][l], nw);
          cvt_f32_f16<<<(nb + 255) / 256, 256, 0, stream>>>(bp, Wb + boff[blk][m][l], nb);
        }
  }

  auto gemm16 = [&](const _Float16* A, const _Float16* Bw, const _Float16* bi,
                    _Float16* C, long Mr, int Nn, int Kk, int lda, int ldb, int ldc,
                    bool relu, const int* mk) {
    dim3 grid((Nn + BNT - 1) / BNT, (unsigned)((Mr + BMT - 1) / BMT), 1);
    if (relu)
      gemm_wmma<false, false, true, false><<<grid, TB, 0, stream>>>(
          A, Bw, bi, nullptr, C, (int)Mr, Nn, Kk, lda, ldb, ldc, 0, 0, 0, 0);
    else if (mk)
      gemm_wmma<false, false, false, true><<<grid, TB, 0, stream>>>(
          A, Bw, bi, mk, C, (int)Mr, Nn, Kk, lda, ldb, ldc, 0, 0, 0, 0);
    else
      gemm_wmma<false, false, false, false><<<grid, TB, 0, stream>>>(
          A, Bw, bi, nullptr, C, (int)Mr, Nn, Kk, lda, ldb, ldc, 0, 0, 0, 0);
  };

  auto run_mlp = [&](int blk, int m, _Float16* X, int ldx, long R,
                     _Float16* h0, _Float16* h1, _Float16* dst, int lddst, const int* mk) {
    gemm16(X, Wb + woff[blk][m][0], Wb + boff[blk][m][0], h0, R, 256, dims[blk][m][0],
           ldx, 256, 256, true, nullptr);
    _Float16* cur = h0; _Float16* nxt = h1;
    for (int l = 1; l <= 3; ++l) {
      gemm16(cur, Wb + woff[blk][m][l], Wb + boff[blk][m][l], nxt, R, 256, 256,
             256, 256, 256, true, nullptr);
      _Float16* t = cur; cur = nxt; nxt = t;
    }
    gemm16(cur, Wb + woff[blk][m][4], Wb + boff[blk][m][4], dst, R, 128, 256,
           256, 128, lddst, false, mk);
  };

  // encoders
  encode_nodes<<<(B * NV + 255) / 256, 256, 0, stream>>>(Va, pValv, pXv, NV, 128, 384);
  encode_edges<<<(B * NE + 255) / 256, 256, 0, stream>>>(Ea, pXe, NE, 512);

  for (int blk = 0; blk < 4; ++blk) {
    const int Ce = blk ? 128 : 6, Cv = blk ? 128 : 24, Cu = blk ? 128 : 0;
    if (blk) {
      long ne = (long)B * NE * 16, nv = (long)B * NV * 16;  // v8h chunks
      copy_cols8<<<(unsigned)((ne + 255) / 256), 256, 0, stream>>>(pVale, pXe, (long)B * NE, 128, 128, 512, 0);
      copy_cols8<<<(unsigned)((nv + 255) / 256), 256, 0, stream>>>(pValv, pXv, (long)B * NV, 128, 128, 384, 0);
    }
    // edge aggregation: X_e[:, Ce..] = Es^T @ val_v ; X_e[:, Ce+Cv..] = Er^T @ val_v
    {
      dim3 grid((Cv + BNT - 1) / BNT, NE / BMT, B);
      gemm_wmma<true, true, false, false><<<grid, TB, 0, stream>>>(
          Es, pValv, nullptr, nullptr, pXe + Ce, NE, Cv, NV, NE, 128, 512,
          (long)NV * NE, (long)NV * 128, (long)NE * 512, 0);
      gemm_wmma<true, true, false, false><<<grid, TB, 0, stream>>>(
          Er, pValv, nullptr, nullptr, pXe + Ce + Cv, NE, Cv, NV, NE, 128, 512,
          (long)NV * NE, (long)NV * 128, (long)NE * 512, 0);
    }
    if (Cu) {
      long n = (long)B * NE * 16;
      bcast_u8<<<(unsigned)((n + 255) / 256), 256, 0, stream>>>(pU, pXe, NE, 512, Ce + 2 * Cv);
    }
    // fe MLP -> out_E (masked)
    run_mlp(blk, 0, pXe, 512, (long)B * NE, pHe0, pXe /*alias*/, pVale, 128, mask_e);

    // node aggregation: X_v[:, Cv..] = Er @ out_E
    {
      dim3 grid(1, NV / BMT, B);
      gemm_wmma<true, false, false, false><<<grid, TB, 0, stream>>>(
          Er, pVale, nullptr, nullptr, pXv + Cv, NV, 128, NE, NE, 128, 384,
          (long)NV * NE, (long)NE * 128, (long)NV * 384, 0);
    }
    if (Cu) {
      long n = (long)B * NV * 16;
      bcast_u8<<<(unsigned)((n + 255) / 256), 256, 0, stream>>>(pU, pXv, NV, 384, Cv + 128);
    }
    // fv MLP -> out_V (masked)
    run_mlp(blk, 1, pXv, 384, (long)B * NV, pHv0, pXv /*alias*/, pValv, 128, mask_v);

    // global features: [u, sum(out_V), sum(out_E)]
    if (Cu)
      copy_cols8<<<(B * 16 + 255) / 256, 256, 0, stream>>>(pU, pXu, B, 128, 128, 384, 0);
    colsum<<<B, 128, 0, stream>>>(pValv, pXu, NV, 128, 384, Cu);
    colsum<<<B, 128, 0, stream>>>(pVale, pXu, NE, 128, 384, Cu + 128);
    // fu MLP -> u
    run_mlp(blk, 2, pXu, 384, 32, pHu0, pHu1, pU, 128, nullptr);
  }

  final_out<<<B, 128, 0, stream>>>(pU, Wout, bout, out);
}
